// StructuredGenerator_12558484373696
// MI455X (gfx1250) — matrix operand order repaired and checked by reading, not compiled, not run on hardware
//
#include <hip/hip_runtime.h>

// ---------------------------------------------------------------------------
// StructuredGenerator for MI455X (gfx1250, wave32, WMMA)
//   B=64, L=256, S=512, H=800
// Pipeline:
//   1) convert all weights fp32 -> f16 (natural [N][K] layout == WMMA B frags)
//   2) persistent GRU scan kernel (grid barrier across 16 WGs, WMMA per step)
//   3) big WMMA GEMMs for LSTM x-gates (register-tiled 4 m-tiles per wave,
//      B-fragment reuse, async global->LDS A staging via inline CDNA5 asm)
//   4) persistent LSTM scan kernels (4 directions, WMMA h@Whh^T per step)
//   5) concat/reverse builders, final projection
// ---------------------------------------------------------------------------

#define B_ 64
#define L_ 256
#define S_ 512
#define H_ 800

typedef _Float16 half_t;
typedef __attribute__((ext_vector_type(16))) _Float16 v16h;
typedef __attribute__((ext_vector_type(8)))  float    v8f;

union F16Frag { v16h v; half_t h[16]; };

// ---- async global -> LDS staging (GLOBAL_LOAD_ASYNC_TO_LDS_B128, ASYNCcnt) ----
// Inline asm per CDNA5 ISA (portable across toolchains;
// the clang builtin's pointer typing is not spellable from HIP source).
#define USE_ASYNC_ASM 1

__device__ __forceinline__ void async_copy16(const half_t* src, half_t* dst) {
#if USE_ASYNC_ASM
  unsigned lds_off = (unsigned)(size_t)(__attribute__((address_space(3))) half_t*)dst;
  unsigned long long ga = (unsigned long long)(size_t)src;
  asm volatile("global_load_async_to_lds_b128 %0, %1, off"
               :: "v"(lds_off), "v"(ga)
               : "memory");
#else
  *(uint4*)dst = *(const uint4*)src;
#endif
}

__device__ __forceinline__ void stage_fence() {
#if USE_ASYNC_ASM
#if defined(__has_builtin) && __has_builtin(__builtin_amdgcn_s_wait_asynccnt)
  __builtin_amdgcn_s_wait_asynccnt(0);
#else
  asm volatile("s_wait_asynccnt 0x0" ::: "memory");
#endif
#endif
  __syncthreads();
}

// Copy rows x cols (cols % 8 == 0) from global (row stride gstride) into LDS
// (row stride sstride), cooperatively across nthr threads, 16B per transfer.
__device__ __forceinline__ void stage_rows(const half_t* __restrict__ gsrc, size_t gstride,
                                           half_t* sdst, int sstride,
                                           int rows, int cols, int tid, int nthr) {
  const int cvec = cols >> 3;
  const int nvec = rows * cvec;
  for (int i = tid; i < nvec; i += nthr) {
    int r = i / cvec;
    int c = (i - r * cvec) << 3;
    async_copy16(gsrc + (size_t)r * gstride + c, sdst + r * sstride + c);
  }
}

__device__ __forceinline__ v8f wmma_f16(v16h a, v16h b, v8f c) {
  return __builtin_amdgcn_wmma_f32_16x16x32_f16(
      /*neg_a=*/false, a, /*neg_b=*/false, b,
      /*c_mod=*/(short)0, c, /*reuse_a=*/false, /*reuse_b=*/false);
}

// A fragment: 16x32 f16, source row-major [m][k], row stride ld.
// lanes 0-15: M=lane, k 0..7 then 16..23; lanes 16-31: M=lane-16, k 8..15, 24..31.
__device__ __forceinline__ v16h load_a_frag(const half_t* base, int ld, int lane) {
  const int m  = lane & 15;
  const int kb = (lane < 16) ? 0 : 8;
  const half_t* row = base + m * ld;
  F16Frag f;
#pragma unroll
  for (int j = 0; j < 8; ++j) f.h[j] = row[kb + j];
#pragma unroll
  for (int j = 0; j < 8; ++j) f.h[8 + j] = row[16 + kb + j];
  return f.v;
}

// B fragment: 32x16 f16 of W^T, W stored row-major [n][k] (natural layout).
// lanes 0-15: n=lane, k 0..7 then 16..23; lanes 16-31: n=lane-16, k 8..15, 24..31.
__device__ __forceinline__ v16h load_b_frag(const half_t* base /* = W + n0*ldk + k0 */,
                                            int ldk, int lane) {
  const int n  = lane & 15;
  const int kb = (lane < 16) ? 0 : 8;
  const half_t* row = base + (size_t)n * ldk + kb;
  F16Frag f;
#pragma unroll
  for (int j = 0; j < 8; ++j) f.h[j] = row[j];
#pragma unroll
  for (int j = 0; j < 8; ++j) f.h[8 + j] = row[16 + j];
  return f.v;
}

__device__ __forceinline__ float fast_sigmoid(float x) {
  return __builtin_amdgcn_rcpf(1.f + __expf(-x));
}
__device__ __forceinline__ float fast_tanh(float x) {
  return 2.f * __builtin_amdgcn_rcpf(1.f + __expf(-2.f * x)) - 1.f;
}

// Epoch-based grid barrier: bar[0]=count, bar[1]=epoch (generation agnostic).
__device__ __forceinline__ void grid_barrier(unsigned* bar, unsigned nwg) {
  __syncthreads();
  if (threadIdx.x == 0) {
    __threadfence();
    volatile unsigned* vb = bar;
    unsigned e = vb[1];
    unsigned a = atomicAdd(&bar[0], 1u);
    if (a == nwg - 1u) {
      atomicExch(&bar[0], 0u);
      __threadfence();
      atomicAdd(&bar[1], 1u);
    } else {
      while (vb[1] == e) { __builtin_amdgcn_s_sleep(1); }
      __threadfence();
    }
  }
  __syncthreads();
}

// ---------------------------------------------------------------------------
// fp32 -> f16 weight conversion
// ---------------------------------------------------------------------------
__global__ void k_f32tof16(const float* __restrict__ in, half_t* __restrict__ out, int n) {
  int i = blockIdx.x * blockDim.x + threadIdx.x;
  if (i < n) out[i] = (half_t)in[i];
}

// ---------------------------------------------------------------------------
// Persistent autoregressive GRU scan.  16 WGs x 256 threads (8 waves).
// ---------------------------------------------------------------------------
#define GRU_NWG 16
#define LDA_G 520  // padded LDS row stride (f16 elems), 1040B = 16B multiple

struct GruP {
  const float* latent;
  const float* noise;
  const half_t* Wih[3];
  const half_t* Whh[3];
  const float* bih[3];
  const float* bhh[3];
  const half_t* Wm;
  const float* bm;
  const half_t* Wl;
  const float* bl;
  half_t* xcur;   // [64][512]
  half_t* h[3];   // [64][512] each
  half_t* xseq;   // [B][L][512]
  unsigned* bar;
};

__global__ __launch_bounds__(256) void k_gru_scan(GruP p) {
  __shared__ __align__(16) half_t sA[16 * LDA_G];
  __shared__ __align__(16) half_t sH[16 * LDA_G];
  const int tid = threadIdx.x, lane = tid & 31, wave = tid >> 5;
  const int wg = blockIdx.x;
  const int m0 = (wg & 3) * 16;
  const int n0 = (((wg >> 2) << 3) + wave) * 16;

  for (int i = wg * 256 + tid; i < B_ * S_; i += GRU_NWG * 256) {
    p.xcur[i] = (half_t)p.latent[i];
    p.h[0][i] = (half_t)0.f;
    p.h[1][i] = (half_t)0.f;
    p.h[2][i] = (half_t)0.f;
  }
  grid_barrier(p.bar, GRU_NWG);

  for (int t = 0; t < L_; ++t) {
    for (int c = 0; c < 3; ++c) {
      const half_t* xin = (c == 0) ? p.xcur : p.h[c - 1];
      const half_t* hin = p.h[c];
      stage_rows(xin + (size_t)m0 * S_, S_, sA, LDA_G, 16, S_, tid, 256);
      stage_rows(hin + (size_t)m0 * S_, S_, sH, LDA_G, 16, S_, tid, 256);
      stage_fence();

      v8f gi0 = {}, gi1 = {}, gi2 = {}, gh0 = {}, gh1 = {}, gh2 = {};
      const half_t* Wi = p.Wih[c];
      const half_t* Wh = p.Whh[c];
      for (int kk = 0; kk < S_; kk += 32) {
        v16h ax = load_a_frag(&sA[kk], LDA_G, lane);
        v16h ah = load_a_frag(&sH[kk], LDA_G, lane);
        v16h b;
        b = load_b_frag(Wi + (size_t)(n0) * S_ + kk, S_, lane);          gi0 = wmma_f16(ax, b, gi0);
        b = load_b_frag(Wi + (size_t)(n0 + S_) * S_ + kk, S_, lane);     gi1 = wmma_f16(ax, b, gi1);
        b = load_b_frag(Wi + (size_t)(n0 + 2 * S_) * S_ + kk, S_, lane); gi2 = wmma_f16(ax, b, gi2);
        b = load_b_frag(Wh + (size_t)(n0) * S_ + kk, S_, lane);          gh0 = wmma_f16(ah, b, gh0);
        b = load_b_frag(Wh + (size_t)(n0 + S_) * S_ + kk, S_, lane);     gh1 = wmma_f16(ah, b, gh1);
        b = load_b_frag(Wh + (size_t)(n0 + 2 * S_) * S_ + kk, S_, lane); gh2 = wmma_f16(ah, b, gh2);
      }

      const int j = n0 + (lane & 15);
      const int mr = (lane >> 4) << 3;
#pragma unroll
      for (int r = 0; r < 8; ++r) {
        int m = m0 + mr + r;
        float ir = gi0[r] + p.bih[c][j];
        float hr = gh0[r] + p.bhh[c][j];
        float iz = gi1[r] + p.bih[c][S_ + j];
        float hz = gh1[r] + p.bhh[c][S_ + j];
        float in_ = gi2[r] + p.bih[c][2 * S_ + j];
        float hn = gh2[r] + p.bhh[c][2 * S_ + j];
        float rg = fast_sigmoid(ir + hr);
        float zg = fast_sigmoid(iz + hz);
        float ng = fast_tanh(in_ + rg * hn);
        float hp = (float)sH[(mr + r) * LDA_G + j];
        float hnew = (1.f - zg) * ng + zg * hp;
        p.h[c][(size_t)m * S_ + j] = (half_t)hnew;
      }
      grid_barrier(p.bar, GRU_NWG);
    }

    // mean/std projection + sample
    {
      stage_rows(p.h[2] + (size_t)m0 * S_, S_, sA, LDA_G, 16, S_, tid, 256);
      stage_fence();

      v8f am = {}, al = {};
      for (int kk = 0; kk < S_; kk += 32) {
        v16h a = load_a_frag(&sA[kk], LDA_G, lane);
        v16h b;
        b = load_b_frag(p.Wm + (size_t)n0 * S_ + kk, S_, lane); am = wmma_f16(a, b, am);
        b = load_b_frag(p.Wl + (size_t)n0 * S_ + kk, S_, lane); al = wmma_f16(a, b, al);
      }

      const int j = n0 + (lane & 15);
      const int mr = (lane >> 4) << 3;
#pragma unroll
      for (int r = 0; r < 8; ++r) {
        int m = m0 + mr + r;
        float mean = am[r] + p.bm[j];
        float sd = __expf((al[r] + p.bl[j]) * 0.5f);
        float z = p.noise[((size_t)m * L_ + t) * S_ + j];
        float samp = z * sd + mean;
        p.xcur[(size_t)m * S_ + j] = (half_t)samp;
        p.xseq[((size_t)m * L_ + t) * S_ + j] = (half_t)samp;
      }
      grid_barrier(p.bar, GRU_NWG);
    }
  }
}

// ---------------------------------------------------------------------------
// Big time-parallel GEMM: Y[M][3200] = X[M][K] @ W[3200][K]^T  (f16 in/out,
// f32 accumulate).  WG = 8 waves, covers 64 A-rows (4 m-tiles staged in LDS
// in K-chunks) x 8 n-tiles (one per wave).  Each wave holds 4 accumulators and
// reuses each B fragment 4x -> global B traffic per WMMA cut 4x.
// ---------------------------------------------------------------------------
#define GEMM_NTB 25           // 3200 / 16 / 8
#define GEMM_KC 256           // K chunk staged in LDS
#define GEMM_LDS 264          // padded stride (528B, 16B multiple)

struct GemmP {
  const half_t* X;
  const half_t* W;
  half_t* Y;
  int K;
};

__global__ __launch_bounds__(256) void k_gemm_xg(GemmP p) {
  __shared__ __align__(16) half_t sA[64 * GEMM_LDS];
  const int tid = threadIdx.x, lane = tid & 31, wave = tid >> 5;
  const int m0 = (blockIdx.x / GEMM_NTB) * 64;
  const int n0 = ((blockIdx.x % GEMM_NTB) * 8 + wave) * 16;
  const int K = p.K;

  v8f acc0 = {}, acc1 = {}, acc2 = {}, acc3 = {};
  const half_t* wb = p.W + (size_t)n0 * K;

  for (int k0 = 0; k0 < K; k0 += GEMM_KC) {
    const int kc = (K - k0 < GEMM_KC) ? (K - k0) : GEMM_KC;
    stage_rows(p.X + (size_t)m0 * K + k0, K, sA, GEMM_LDS, 64, kc, tid, 256);
    stage_fence();

    for (int kk = 0; kk < kc; kk += 32) {
      __builtin_prefetch((const void*)(wb + (size_t)(lane & 15) * K + k0 + kk + 64), 0, 3);
      v16h b = load_b_frag(wb + k0 + kk, K, lane);
      v16h a0 = load_a_frag(&sA[0 * 16 * GEMM_LDS + kk], GEMM_LDS, lane);
      v16h a1 = load_a_frag(&sA[1 * 16 * GEMM_LDS + kk], GEMM_LDS, lane);
      v16h a2 = load_a_frag(&sA[2 * 16 * GEMM_LDS + kk], GEMM_LDS, lane);
      v16h a3 = load_a_frag(&sA[3 * 16 * GEMM_LDS + kk], GEMM_LDS, lane);
      acc0 = wmma_f16(a0, b, acc0);
      acc1 = wmma_f16(a1, b, acc1);
      acc2 = wmma_f16(a2, b, acc2);
      acc3 = wmma_f16(a3, b, acc3);
    }
    __syncthreads();  // protect sA before next chunk staging
  }

  const int j = n0 + (lane & 15);
  const int mr = (lane >> 4) << 3;
#pragma unroll
  for (int r = 0; r < 8; ++r) {
    p.Y[(size_t)(m0 + 0 * 16 + mr + r) * 3200 + j] = (half_t)acc0[r];
    p.Y[(size_t)(m0 + 1 * 16 + mr + r) * 3200 + j] = (half_t)acc1[r];
    p.Y[(size_t)(m0 + 2 * 16 + mr + r) * 3200 + j] = (half_t)acc2[r];
    p.Y[(size_t)(m0 + 3 * 16 + mr + r) * 3200 + j] = (half_t)acc3[r];
  }
}

// ---------------------------------------------------------------------------
// Persistent LSTM scan (one direction).  28 WGs x 256 threads.
// ---------------------------------------------------------------------------
#define LSTM_NWG 28
#define LDH_L 808  // 1616B row stride, 16B multiple

struct LstmP {
  const half_t* xg;    // [B][L][3200], precomputed x@Wih^T
  const half_t* Whh;   // [3200][800] natural row-major
  const float* bih;
  const float* bhh;
  const int* len;
  half_t* hbuf;        // [64][800]
  float* cbuf;         // [64][800]
  half_t* y;           // [B][L][800] scan-order outputs
  unsigned* bar;
  int dir;             // 0 fwd, 1 bwd (index reversed-padded xg rows)
};

__global__ __launch_bounds__(256) void k_lstm_scan(LstmP p) {
  __shared__ __align__(16) half_t sH[16 * LDH_L];
  const int tid = threadIdx.x, lane = tid & 31, wave = tid >> 5;
  const int wg = blockIdx.x;
  const int m0 = (wg & 3) * 16;
  const int nt = (wg >> 2) * 8 + wave;
  const int n0 = nt * 16;
  const bool act = nt < 50;

  for (int i = wg * 256 + tid; i < B_ * H_; i += LSTM_NWG * 256) {
    p.hbuf[i] = (half_t)0.f;
    p.cbuf[i] = 0.f;
  }
  grid_barrier(p.bar, LSTM_NWG);

  for (int t = 0; t < L_; ++t) {
    stage_rows(p.hbuf + (size_t)m0 * H_, H_, sH, LDH_L, 16, H_, tid, 256);
    stage_fence();

    if (act) {
      v8f ai = {}, af = {}, ag = {}, ao = {};
      for (int kk = 0; kk < H_; kk += 32) {
        __builtin_prefetch((const void*)(p.Whh + (size_t)(n0 + (lane & 15)) * H_ + kk + 64), 0, 3);
        v16h a = load_a_frag(&sH[kk], LDH_L, lane);
        v16h b;
        b = load_b_frag(p.Whh + (size_t)(n0) * H_ + kk, H_, lane);          ai = wmma_f16(a, b, ai);
        b = load_b_frag(p.Whh + (size_t)(n0 + H_) * H_ + kk, H_, lane);     af = wmma_f16(a, b, af);
        b = load_b_frag(p.Whh + (size_t)(n0 + 2 * H_) * H_ + kk, H_, lane); ag = wmma_f16(a, b, ag);
        b = load_b_frag(p.Whh + (size_t)(n0 + 3 * H_) * H_ + kk, H_, lane); ao = wmma_f16(a, b, ao);
      }

      const int j = n0 + (lane & 15);
      const int mr = (lane >> 4) << 3;
#pragma unroll
      for (int r = 0; r < 8; ++r) {
        int m = m0 + mr + r;
        int lenb = p.len[m];
        int xrow;
        if (p.dir) {
          int rr = lenb - 1 - t;
          xrow = rr < 0 ? 0 : (rr > L_ - 1 ? L_ - 1 : rr);
        } else {
          xrow = t;
        }
        const half_t* xr = p.xg + ((size_t)m * L_ + xrow) * 3200;
        float gi = ai[r] + (float)xr[j] + p.bih[j] + p.bhh[j];
        float gf = af[r] + (float)xr[H_ + j] + p.bih[H_ + j] + p.bhh[H_ + j];
        float gg = ag[r] + (float)xr[2 * H_ + j] + p.bih[2 * H_ + j] + p.bhh[2 * H_ + j];
        float go = ao[r] + (float)xr[3 * H_ + j] + p.bih[3 * H_ + j] + p.bhh[3 * H_ + j];
        float co = p.cbuf[(size_t)m * H_ + j];
        float cn = fast_sigmoid(gf) * co + fast_sigmoid(gi) * fast_tanh(gg);
        float hn = fast_sigmoid(go) * fast_tanh(cn);
        bool msk = t < lenb;
        float hp = (float)sH[(mr + r) * LDH_L + j];
        p.cbuf[(size_t)m * H_ + j] = msk ? cn : co;
        p.hbuf[(size_t)m * H_ + j] = (half_t)(msk ? hn : hp);
        p.y[((size_t)m * L_ + t) * H_ + j] = (half_t)(msk ? hn : 0.f);
      }
    }
    grid_barrier(p.bar, LSTM_NWG);
  }
}

// ---------------------------------------------------------------------------
// Build [B][L][1600] = concat(yf, reverse_padded(yb) * mask)
// ---------------------------------------------------------------------------
__global__ void k_concat_rev(const half_t* __restrict__ yf, const half_t* __restrict__ yb,
                             const int* __restrict__ len, half_t* __restrict__ out, int n) {
  int i = blockIdx.x * blockDim.x + threadIdx.x;
  if (i >= n) return;
  int j = i % 1600;
  int bt = i / 1600;
  int t = bt % L_;
  int b = bt / L_;
  int lenb = len[b];
  half_t v;
  if (j < H_) {
    v = yf[((size_t)b * L_ + t) * H_ + j];
  } else {
    int rr = lenb - 1 - t;
    rr = rr < 0 ? 0 : (rr > L_ - 1 ? L_ - 1 : rr);
    float vb = (t < lenb) ? (float)yb[((size_t)b * L_ + rr) * H_ + (j - H_)] : 0.f;
    v = (half_t)vb;
  }
  out[i] = v;
}

// ---------------------------------------------------------------------------
// Final projection: angles[b][t][a] = (h2 . Wa[a]) + ba[a], masked.
// ---------------------------------------------------------------------------
__global__ void k_proj(const half_t* __restrict__ H2, const float* __restrict__ Wa,
                       const float* __restrict__ ba, const int* __restrict__ len,
                       float* __restrict__ out) {
  int i = blockIdx.x * blockDim.x + threadIdx.x;
  if (i >= B_ * L_ * 3) return;
  int a = i % 3;
  int bt = i / 3;
  int t = bt % L_;
  int b = bt / L_;
  const half_t* h = H2 + (size_t)bt * 1600;
  const float* w = Wa + (size_t)a * 1600;
  float s = 0.f;
  for (int k = 0; k < 1600; ++k) s += (float)h[k] * w[k];
  s += ba[a];
  out[i] = (t < len[b]) ? s : 0.f;
}

// ---------------------------------------------------------------------------
// Host launcher
// ---------------------------------------------------------------------------
extern "C" void kernel_launch(void* const* d_in, const int* in_sizes, int n_in,
                              void* d_out, int out_size, void* d_ws, size_t ws_size,
                              hipStream_t stream) {
  (void)in_sizes; (void)n_in; (void)out_size; (void)ws_size;

  // ---- inputs (setup_inputs dict order) ----
  const float* latent = (const float*)d_in[0];
  const float* noise  = (const float*)d_in[1];
  const int*   lengths = (const int*)d_in[2];
  const float* gWih[3]; const float* gWhh[3]; const float* gbih[3]; const float* gbhh[3];
  int ii = 3;
  for (int c = 0; c < 3; ++c) {
    gWih[c] = (const float*)d_in[ii++];
    gWhh[c] = (const float*)d_in[ii++];
    gbih[c] = (const float*)d_in[ii++];
    gbhh[c] = (const float*)d_in[ii++];
  }
  const float* Wm = (const float*)d_in[ii++];
  const float* bm = (const float*)d_in[ii++];
  const float* Wl = (const float*)d_in[ii++];
  const float* bl = (const float*)d_in[ii++];
  const float* lW_ih[4]; const float* lW_hh[4]; const float* lb_ih[4]; const float* lb_hh[4];
  for (int c = 0; c < 4; ++c) {  // l0f, l0b, l1f, l1b
    lW_ih[c] = (const float*)d_in[ii++];
    lW_hh[c] = (const float*)d_in[ii++];
    lb_ih[c] = (const float*)d_in[ii++];
    lb_hh[c] = (const float*)d_in[ii++];
  }
  const float* Wa = (const float*)d_in[ii++];
  const float* ba = (const float*)d_in[ii++];

  // ---- workspace carve ----
  char* wsb = (char*)d_ws;
  size_t off = 0;
  auto alloc = [&](size_t bytes) -> void* {
    void* ptr = wsb + off;
    off += (bytes + 255) & ~(size_t)255;
    return ptr;
  };

  half_t* gWih16[3]; half_t* gWhh16[3];
  for (int c = 0; c < 3; ++c) {
    gWih16[c] = (half_t*)alloc((size_t)1536 * 512 * 2);
    gWhh16[c] = (half_t*)alloc((size_t)1536 * 512 * 2);
  }
  half_t* Wm16 = (half_t*)alloc((size_t)512 * 512 * 2);
  half_t* Wl16 = (half_t*)alloc((size_t)512 * 512 * 2);
  half_t* lWih16[4]; half_t* lWhh16[4];
  const size_t ihElems[4] = { (size_t)3200 * 512, (size_t)3200 * 512,
                              (size_t)3200 * 1600, (size_t)3200 * 1600 };
  for (int c = 0; c < 4; ++c) {
    lWih16[c] = (half_t*)alloc(ihElems[c] * 2);
    lWhh16[c] = (half_t*)alloc((size_t)3200 * 800 * 2);
  }
  half_t* xcur = (half_t*)alloc((size_t)B_ * S_ * 2);
  half_t* hst[3];
  for (int c = 0; c < 3; ++c) hst[c] = (half_t*)alloc((size_t)B_ * S_ * 2);
  half_t* xseq = (half_t*)alloc((size_t)B_ * L_ * S_ * 2);
  half_t* xgf  = (half_t*)alloc((size_t)B_ * L_ * 3200 * 2);
  half_t* xgb  = (half_t*)alloc((size_t)B_ * L_ * 3200 * 2);
  half_t* hbuf = (half_t*)alloc((size_t)B_ * H_ * 2);
  float*  cbuf = (float*)alloc((size_t)B_ * H_ * 4);
  half_t* yf  = (half_t*)alloc((size_t)B_ * L_ * H_ * 2);
  half_t* yb  = (half_t*)alloc((size_t)B_ * L_ * H_ * 2);
  half_t* yf2 = (half_t*)alloc((size_t)B_ * L_ * H_ * 2);
  half_t* yb2 = (half_t*)alloc((size_t)B_ * L_ * H_ * 2);
  half_t* Hc1 = (half_t*)alloc((size_t)B_ * L_ * 1600 * 2);
  half_t* Hc2 = (half_t*)alloc((size_t)B_ * L_ * 1600 * 2);
  unsigned* bar = (unsigned*)alloc(256);

  // ---- 1) weight conversion ----
  auto cvt = [&](const float* src, half_t* dst, size_t n) {
    k_f32tof16<<<(unsigned)((n + 255) / 256), 256, 0, stream>>>(src, dst, (int)n);
  };
  for (int c = 0; c < 3; ++c) {
    cvt(gWih[c], gWih16[c], (size_t)1536 * 512);
    cvt(gWhh[c], gWhh16[c], (size_t)1536 * 512);
  }
  cvt(Wm, Wm16, (size_t)512 * 512);
  cvt(Wl, Wl16, (size_t)512 * 512);
  for (int c = 0; c < 4; ++c) {
    cvt(lW_ih[c], lWih16[c], ihElems[c]);
    cvt(lW_hh[c], lWhh16[c], (size_t)3200 * 800);
  }
  (void)hipMemsetAsync(bar, 0, 256, stream);

  // ---- 2) autoregressive GRU scan ----
  GruP gp;
  gp.latent = latent; gp.noise = noise;
  for (int c = 0; c < 3; ++c) {
    gp.Wih[c] = gWih16[c]; gp.Whh[c] = gWhh16[c];
    gp.bih[c] = gbih[c];   gp.bhh[c] = gbhh[c];
  }
  gp.Wm = Wm16; gp.bm = bm; gp.Wl = Wl16; gp.bl = bl;
  gp.xcur = xcur;
  for (int c = 0; c < 3; ++c) gp.h[c] = hst[c];
  gp.xseq = xseq; gp.bar = bar;
  k_gru_scan<<<GRU_NWG, 256, 0, stream>>>(gp);

  // ---- 3) layer-0 x-gate GEMMs (M=16384, N=3200, K=512) ----
  const int M = B_ * L_;
  {
    GemmP g0{xseq, lWih16[0], xgf, S_};
    k_gemm_xg<<<(M / 64) * GEMM_NTB, 256, 0, stream>>>(g0);
    GemmP g1{xseq, lWih16[1], xgb, S_};
    k_gemm_xg<<<(M / 64) * GEMM_NTB, 256, 0, stream>>>(g1);
  }

  // ---- 4) layer-0 scans ----
  {
    LstmP s0{xgf, lWhh16[0], lb_ih[0], lb_hh[0], lengths, hbuf, cbuf, yf, bar, 0};
    k_lstm_scan<<<LSTM_NWG, 256, 0, stream>>>(s0);
    LstmP s1{xgb, lWhh16[1], lb_ih[1], lb_hh[1], lengths, hbuf, cbuf, yb, bar, 1};
    k_lstm_scan<<<LSTM_NWG, 256, 0, stream>>>(s1);
  }

  // ---- 5) h1 = concat(yf, rev(yb)*mask) ----
  {
    int n = B_ * L_ * 1600;
    k_concat_rev<<<(n + 255) / 256, 256, 0, stream>>>(yf, yb, lengths, Hc1, n);
  }

  // ---- 6) layer-1 x-gate GEMMs (K=1600), reuse xgf/xgb ----
  {
    GemmP g0{Hc1, lWih16[2], xgf, 1600};
    k_gemm_xg<<<(M / 64) * GEMM_NTB, 256, 0, stream>>>(g0);
    GemmP g1{Hc1, lWih16[3], xgb, 1600};
    k_gemm_xg<<<(M / 64) * GEMM_NTB, 256, 0, stream>>>(g1);
  }

  // ---- 7) layer-1 scans ----
  {
    LstmP s0{xgf, lWhh16[2], lb_ih[2], lb_hh[2], lengths, hbuf, cbuf, yf2, bar, 0};
    k_lstm_scan<<<LSTM_NWG, 256, 0, stream>>>(s0);
    LstmP s1{xgb, lWhh16[3], lb_ih[3], lb_hh[3], lengths, hbuf, cbuf, yb2, bar, 1};
    k_lstm_scan<<<LSTM_NWG, 256, 0, stream>>>(s1);
  }

  // ---- 8) h2 = concat(yf2, rev(yb2)*mask) ----
  {
    int n = B_ * L_ * 1600;
    k_concat_rev<<<(n + 255) / 256, 256, 0, stream>>>(yf2, yb2, lengths, Hc2, n);
  }

  // ---- 9) projection ----
  {
    int n = B_ * L_ * 3;
    k_proj<<<(n + 255) / 256, 256, 0, stream>>>(Hc2, Wa, ba, lengths, (float*)d_out);
  }
}
